// CVXPolicy_Quadcopter_14216341749944
// MI455X (gfx1250) — matrix operand order repaired and checked by reading, not compiled, not run on hardware
//
#include <hip/hip_runtime.h>
#include <math.h>

// CVX quadcopter policy: MLP(13->100 tanh ->12) then closed-form argmin via
// Lambert W.  B=500000 rows.  Both GEMMs run TRANSPOSED on WMMA:
//   L1: h^T = W1^T(A, 7 M-tiles) x inp^T(B, K=32)   -> 7 WMMAs / 16 rows
//   L2: q^T = W2^T(A, K=128)     x h^T(B, 4 chunks) -> 4 WMMAs / 16 rows
// Hidden columns are PERMUTED across the A-fragments so each lane's layer-1
// D registers are already the layer-2 B operand (no cross-lane exchange).
// Biases fold in as constant-1 K entries (inp[13]=1, h[100]=1).
// TWO row-tiles per iteration (one row per lane) so loads, packing,
// Lambert-W and stores run on the full wave; tanh is the CDNA5 HW TRANS op.

typedef __attribute__((ext_vector_type(16))) _Float16 v16h;
typedef __attribute__((ext_vector_type(2)))  _Float16 h2;
typedef __attribute__((ext_vector_type(8)))  float    v8f;
typedef __attribute__((ext_vector_type(8)))  unsigned v8u;

#define WAVES_PER_BLOCK 8
#define MASS_INV 2.0f   // 1/0.5

static __device__ __forceinline__ float fast_tanh(float x) {
#if __has_builtin(__builtin_amdgcn_tanhf)
    return __builtin_amdgcn_tanhf(x);
#elif __has_builtin(__builtin_amdgcn_tanh_f32)
    return __builtin_amdgcn_tanh_f32(x);
#else
    float y;
    asm volatile("v_tanh_f32 %0, %1\n\tv_nop" : "=v"(y) : "v"(x));
    return y;
#endif
}

static __device__ __forceinline__ unsigned pack2h(float a, float b) {
    h2 p; p.x = (_Float16)a; p.y = (_Float16)b;   // -> v_cvt_pk_f16_f32
    return __builtin_bit_cast(unsigned, p);
}

// lane i <- lane i^16 (half-wave swap), VALU permlane when available
static __device__ __forceinline__ unsigned swap16(unsigned v) {
#if __has_builtin(__builtin_amdgcn_permlanex16)
    return __builtin_amdgcn_permlanex16(v, v, 0x76543210u, 0xfedcba98u,
                                        false, false);
#else
    return (unsigned)__shfl_xor((int)v, 16, 32);
#endif
}

__global__ __launch_bounds__(256)
void cvx_policy_kernel(const float* __restrict__ z,
                       const float* __restrict__ t,
                       const float* __restrict__ W1,   // [13,100] row-major
                       const float* __restrict__ b1,   // [100]
                       const float* __restrict__ W2,   // [100,12] row-major
                       const float* __restrict__ b2,   // [12]
                       float* __restrict__ out,        // [B,4]
                       int B, int tiles32)
{
    // ---- LDS weight staging, already in WMMA A-fragment order ----------
    __shared__ __align__(32) _Float16 sW1f[7 * 32 * 16];
    __shared__ __align__(32) _Float16 sW2f[4 * 32 * 16];

    const int tid  = threadIdx.x;
    const int wave = tid >> 5;
    const int lane = tid & 31;
    const int m    = lane & 15;
    const bool isHi = (lane >= 16);

    for (int i = tid; i < 7 * 32 * 16; i += 256) {
        int e  = i & 15;
        int ln = (i >> 4) & 31;
        int nt = i >> 9;
        int hf = ln >> 4, mi = ln & 15;
        int K   = 16 * hf + e;
        int col = 32 * (nt >> 1) + 8 * (nt & 1) + mi + (mi & 8);
        float v = 0.0f;
        if (col < 100) {
            if (K < 13)       v = W1[K * 100 + col];
            else if (K == 13) v = b1[col];
        }
        sW1f[i] = (_Float16)v;
    }
    for (int i = tid; i < 4 * 32 * 16; i += 256) {
        int e  = i & 15;
        int ln = (i >> 4) & 31;
        int s  = i >> 9;
        int hf = ln >> 4, j = ln & 15;
        int K  = 32 * s + 16 * hf + e;
        float v = 0.0f;
        if (j < 6) {
            if (K < 100)       v = W2[K * 12 + 6 + j];
            else if (K == 100) v = b2[6 + j];
        }
        sW2f[i] = (_Float16)v;
    }
    __syncthreads();

    // ---- per-wave A fragments: contiguous 32B vector loads -------------
    v16h aW1[7];
    #pragma unroll
    for (int nt = 0; nt < 7; ++nt)
        aW1[nt] = ((const v16h*)sW1f)[nt * 32 + lane];
    v16h aW2[4];
    #pragma unroll
    for (int s = 0; s < 4; ++s)
        aW2[s] = ((const v16h*)sW2f)[s * 32 + lane];

    const int waveGlobal = blockIdx.x * WAVES_PER_BLOCK + wave;
    const int nWaves     = gridDim.x * WAVES_PER_BLOCK;

    for (int g = waveGlobal; g < tiles32; g += nWaves) {
        // ---- every lane loads & packs its OWN row (coalesced) ----------
        int row  = g * 32 + lane;
        int rowc = row < B ? row : B - 1;
        const float4* zr = (const float4*)(z + (long)rowc * 12);
        float  tv = t[rowc];
        float4 f0 = zr[0], f1 = zr[1], f2 = zr[2];
        unsigned pk[7];
        pk[0] = pack2h(tv,   f0.x);
        pk[1] = pack2h(f0.y, f0.z);
        pk[2] = pack2h(f0.w, f1.x);
        pk[3] = pack2h(f1.y, f1.z);
        pk[4] = pack2h(f1.w, f2.x);
        pk[5] = pack2h(f2.y, f2.z);
        pk[6] = pack2h(f2.w, 1.0f);        // dim 13 = 1 (bias row of W1)

        // B operands of layer 1 for the two 16-row tiles.  Lanes 0-15
        // carry K=0..15; lanes 16-31 carry K=16..31 (structurally zero).
        v8u bAu = {}, bBu = {};
        #pragma unroll
        for (int j2 = 0; j2 < 7; ++j2) {
            unsigned x = swap16(pk[j2]);   // high-lane rows -> low lanes
            bAu[j2] = isHi ? 0u : pk[j2];
            bBu[j2] = isHi ? 0u : x;
        }
        v16h bInA = __builtin_bit_cast(v16h, bAu);
        v16h bInB = __builtin_bit_cast(v16h, bBu);

        // ---- fused layers, dual-tile ILP -------------------------------
        v8f acc2a = {}, acc2b = {};
        #pragma unroll
        for (int s = 0; s < 4; ++s) {
            v8f h0a = {}, h0b = {}, h1a = {}, h1b = {};
            h0a = __builtin_amdgcn_wmma_f32_16x16x32_f16(
                false, aW1[2 * s], false, bInA, (short)0, h0a, false, false);
            h0b = __builtin_amdgcn_wmma_f32_16x16x32_f16(
                false, aW1[2 * s], false, bInB, (short)0, h0b, false, false);
            if (2 * s + 1 < 7) {          // tile 7 = all-zero pad cols
                h1a = __builtin_amdgcn_wmma_f32_16x16x32_f16(
                    false, aW1[2 * s + 1], false, bInA, (short)0, h1a, false, false);
                h1b = __builtin_amdgcn_wmma_f32_16x16x32_f16(
                    false, aW1[2 * s + 1], false, bInB, (short)0, h1b, false, false);
            }
            v8u buA, buB;
            #pragma unroll
            for (int jp = 0; jp < 4; ++jp) {
                float aa = fast_tanh(h0a[2 * jp]);
                float ab = fast_tanh(h0a[2 * jp + 1]);
                float ba = fast_tanh(h0b[2 * jp]);
                float bb = fast_tanh(h0b[2 * jp + 1]);
                if (s == 3 && jp == 2) {  // col 100 = 1 (bias row of W2)
                    aa = isHi ? aa : 1.0f;
                    ba = isHi ? ba : 1.0f;
                }
                buA[jp] = pack2h(aa, ab);
                buB[jp] = pack2h(ba, bb);
            }
            #pragma unroll
            for (int jp = 0; jp < 4; ++jp) {
                if (2 * s + 1 < 7) {
                    buA[4 + jp] = pack2h(fast_tanh(h1a[2 * jp]),
                                         fast_tanh(h1a[2 * jp + 1]));
                    buB[4 + jp] = pack2h(fast_tanh(h1b[2 * jp]),
                                         fast_tanh(h1b[2 * jp + 1]));
                } else {
                    buA[4 + jp] = 0u;
                    buB[4 + jp] = 0u;
                }
            }
            acc2a = __builtin_amdgcn_wmma_f32_16x16x32_f16(
                false, aW2[s], false, __builtin_bit_cast(v16h, buA),
                (short)0, acc2a, false, false);
            acc2b = __builtin_amdgcn_wmma_f32_16x16x32_f16(
                false, aW2[s], false, __builtin_bit_cast(v16h, buB),
                (short)0, acc2b, false, false);
        }

        // ---- merge q's so every lane owns its row ----------------------
        // acc2a lanes<16: q_j[row g*32+L]; acc2b lanes<16: q_j[row g*32+16+L]
        float q[6];
        #pragma unroll
        for (int j = 0; j < 6; ++j) {
            unsigned xb = swap16(__builtin_bit_cast(unsigned, acc2b[j]));
            float    xf = __builtin_bit_cast(float, xb);
            q[j] = isHi ? xf : acc2a[j];
        }

        // ---- Lambert W + output on ALL 32 lanes ------------------------
        float c0 = (q[0] + q[1] + q[2]) * MASS_INV;
        float c1 = q[3];
        float c2 = q[4];
        float c3 = q[5];
        float x  = c0 * c0 + c1 * c1 + c2 * c2 + c3 * c3;
        float w  = log1pf(x);
        #pragma unroll
        for (int it = 0; it < 30; ++it) {
            float ew = __expf(w);                              // v_exp_f32
            float d  = __builtin_amdgcn_rcpf(ew * (w + 1.0f)); // v_rcp_f32
            w = w - (w * ew - x) * d;
        }
        float sc = -__expf(-0.5f * w);
        if (row < B) {
            float4 u = make_float4(c0 * sc, c1 * sc, c2 * sc, c3 * sc);
            *(float4*)(out + (long)row * 4) = u;
        }
    }
}

extern "C" void kernel_launch(void* const* d_in, const int* in_sizes, int n_in,
                              void* d_out, int out_size, void* d_ws, size_t ws_size,
                              hipStream_t stream) {
    const float* z  = (const float*)d_in[0];
    const float* t  = (const float*)d_in[1];
    const float* W1 = (const float*)d_in[2];
    const float* b1 = (const float*)d_in[3];
    const float* W2 = (const float*)d_in[4];
    const float* b2 = (const float*)d_in[5];
    float* out = (float*)d_out;

    int B = in_sizes[1];                 // t has one entry per batch row
    int tiles32 = (B + 31) / 32;         // 32 rows per wave-iteration
    int blocks = (tiles32 + WAVES_PER_BLOCK - 1) / WAVES_PER_BLOCK;
    if (blocks > 1024) blocks = 1024;    // persistent waves, grid-stride
    cvx_policy_kernel<<<blocks, 256, 0, stream>>>(z, t, W1, b1, W2, b2, out,
                                                  B, tiles32);
}
